// VQBottleneckWrapper_36412732735591
// MI455X (gfx1250) — compile-verified, matrix operand order checked by reading, not run on hardware
//
#include <hip/hip_runtime.h>
#include <hip/hip_bf16.h>

// ---------------------------------------------------------------------------
// VQ bottleneck for MI455X (gfx1250, wave32, WMMA).
// Distances via v_wmma_f32_16x16x32_bf16; global->LDS staging via CDNA5
// async-to-LDS instructions (ASYNCcnt) with a double-buffered pipeline.
// Waves tile the 128x128 block as a 4x2 grid of 32x64 tiles (1.5 ds/wmma).
// ---------------------------------------------------------------------------

typedef __attribute__((ext_vector_type(16))) __bf16 v16bf;
typedef __attribute__((ext_vector_type(8)))  __bf16 v8bf;
typedef __attribute__((ext_vector_type(8)))  float  v8f;

#define BT   16384   // B*T = 4*4096
#define DDIM 1024
#define KCB  8192
#define BM   128
#define BN   128
#define BK   128
#define NCHUNK ((KCB / BN) * (DDIM / BK))   // 64 * 8 = 512 pipeline steps

#if defined(__has_builtin)
#if __has_builtin(__builtin_amdgcn_sched_group_barrier)
#define SCHED_GROUP(mask, size, id) __builtin_amdgcn_sched_group_barrier(mask, size, id)
#endif
#endif
#ifndef SCHED_GROUP
#define SCHED_GROUP(mask, size, id)
#endif

__device__ __forceinline__ unsigned short f2bf(float f) {
    unsigned u = __float_as_uint(f);
    unsigned r = u + 0x7FFFu + ((u >> 16) & 1u);   // round-to-nearest-even
    return (unsigned short)(r >> 16);
}

__device__ __forceinline__ void async_copy_b128(unsigned ldsByteOff, const void* gaddr) {
    // GLOBAL_LOAD_ASYNC_TO_LDS_B128: per-lane 16B global->LDS, ASYNCcnt-tracked.
    asm volatile("global_load_async_to_lds_b128 %0, %1, off"
                 :: "v"(ldsByteOff), "v"(gaddr) : "memory");
}

__device__ __forceinline__ void wait_async0() {
    asm volatile("s_wait_asynccnt 0x0" ::: "memory");
}

// ---- zero the scalar accumulators (must happen every launch) --------------
__global__ void vq_init(float* accum) {
    if (threadIdx.x < 2) accum[threadIdx.x] = 0.0f;
}

// ---- convert h [BT*DDIM] f32 -> bf16 --------------------------------------
__global__ __launch_bounds__(256) void vq_cvt_h(const float* __restrict__ h,
                                                unsigned short* __restrict__ hbf) {
    int i = (blockIdx.x * 256 + threadIdx.x) * 4;
    float4 v = *(const float4*)(h + i);
    ushort4 o;
    o.x = f2bf(v.x); o.y = f2bf(v.y); o.z = f2bf(v.z); o.w = f2bf(v.w);
    *(ushort4*)(hbf + i) = o;
}

// ---- convert codebook row -> bf16 and compute c2[row] ---------------------
__global__ __launch_bounds__(256) void vq_cvt_cb(const float* __restrict__ cb,
                                                 unsigned short* __restrict__ cbbf,
                                                 float* __restrict__ c2) {
    __shared__ float red[256];
    int row = blockIdx.x;
    int t = threadIdx.x;
    const float* p = cb + (size_t)row * DDIM + t * 4;
    float4 v = *(const float4*)p;
    ushort4 o;
    o.x = f2bf(v.x); o.y = f2bf(v.y); o.z = f2bf(v.z); o.w = f2bf(v.w);
    *(ushort4*)(cbbf + (size_t)row * DDIM + t * 4) = o;
    red[t] = v.x * v.x + v.y * v.y + v.z * v.z + v.w * v.w;
    __syncthreads();
    for (int s = 128; s > 0; s >>= 1) {
        if (t < s) red[t] += red[t + s];
        __syncthreads();
    }
    if (t == 0) c2[row] = red[0];
}

// ---- main WMMA distance + argmin kernel -----------------------------------
__global__ __launch_bounds__(256) void vq_argmin_wmma(
        const unsigned short* __restrict__ hbf,
        const unsigned short* __restrict__ cbbf,
        const float* __restrict__ c2,
        int* __restrict__ idx_out,
        float* __restrict__ idxf_out) {
    extern __shared__ unsigned short smem[];   // 2 buffers x (A 32KB | B 32KB)

    const int tid  = threadIdx.x;
    const int wave = tid >> 5;
    const int lane = tid & 31;
    const int lo   = lane & 15;
    const int hi   = lane >> 4;
    const int wr   = wave & 3;    // row strip: rows wr*32 .. wr*32+31
    const int wc   = wave >> 2;   // col strip: cols wc*64 .. wc*64+63
    const int rowBase = blockIdx.x * BM;

    // low 32 bits of a shared-aperture generic address == LDS byte offset
    const unsigned sbase = (unsigned)(size_t)(void*)smem;

    // per-thread staging coordinates (16B units; 16 units per 128-elem row)
    const int srow = tid >> 4;           // 0..15  (strided by 16 below)
    const int skv  = (tid & 15) << 3;    // element offset within row

    float minv[2][8];
    int   mini[2][8];
#pragma unroll
    for (int m = 0; m < 2; ++m)
#pragma unroll
        for (int r = 0; r < 8; ++r) { minv[m][r] = 3.4e38f; mini[m][r] = 0; }

    v8f acc[2][4];
#pragma unroll
    for (int m = 0; m < 2; ++m)
#pragma unroll
        for (int n = 0; n < 4; ++n)
#pragma unroll
            for (int r = 0; r < 8; ++r) acc[m][n][r] = 0.0f;

    // ---- async staging of chunk s into buffer (s&1) ----
    auto issue = [&](int s) {
        const int nt = (s >> 3) * BN;
        const int kc = (s & 7) * BK;
        const unsigned bufB = sbase + (unsigned)(s & 1) * 65536u;
        // A: BM x BK  (rows srow, srow+16, ... srow+112)
#pragma unroll
        for (int j = 0; j < 8; ++j) {
            const int row = srow + j * 16;
            async_copy_b128(bufB + (unsigned)(row * BK + skv) * 2u,
                            hbf + (size_t)(rowBase + row) * DDIM + kc + skv);
        }
        // B: BN x BK at +32768 bytes
#pragma unroll
        for (int j = 0; j < 8; ++j) {
            const int col = srow + j * 16;
            async_copy_b128(bufB + 32768u + (unsigned)(col * BK + skv) * 2u,
                            cbbf + (size_t)(nt + col) * DDIM + kc + skv);
        }
    };

    issue(0);

    for (int s = 0; s < NCHUNK; ++s) {
        wait_async0();        // my loads for buffer (s&1) complete
        __syncthreads();      // everyone's complete; everyone done with other buf
        if (s + 1 < NCHUNK) issue(s + 1);   // overlap next stage with compute

        const unsigned short* As = smem + (s & 1) * 65536 / 2;
        const unsigned short* Bs = As + 32768 / 2;

#pragma unroll
        for (int kk = 0; kk < BK; kk += 32) {
            // 2 A fragments (rows wr*32 + 16m) + 4 B fragments (cols wc*64 + 16n)
            v16bf av[2];
#pragma unroll
            for (int m = 0; m < 2; ++m) {
                const unsigned short* ap =
                    &As[(wr * 32 + m * 16 + lo) * BK + kk + hi * 8];
                v8bf a0 = *(const v8bf*)ap;
                v8bf a1 = *(const v8bf*)(ap + 16);
                av[m] = __builtin_shufflevector(a0, a1,
                    0, 1, 2, 3, 4, 5, 6, 7, 8, 9, 10, 11, 12, 13, 14, 15);
            }
            v16bf bv[4];
#pragma unroll
            for (int n = 0; n < 4; ++n) {
                const unsigned short* bp =
                    &Bs[(wc * 64 + n * 16 + lo) * BK + kk + hi * 8];
                v8bf b0 = *(const v8bf*)bp;
                v8bf b1 = *(const v8bf*)(bp + 16);
                bv[n] = __builtin_shufflevector(b0, b1,
                    0, 1, 2, 3, 4, 5, 6, 7, 8, 9, 10, 11, 12, 13, 14, 15);
            }
#pragma unroll
            for (int m = 0; m < 2; ++m)
#pragma unroll
                for (int n = 0; n < 4; ++n)
                    acc[m][n] = __builtin_amdgcn_wmma_f32_16x16x32_bf16(
                        false, av[m], false, bv[n], (short)0, acc[m][n], false, false);
            // schedule: batch the 12 LDS reads, then the 8 WMMAs back-to-back
            SCHED_GROUP(0x100, 12, 0);   // DS read group
            SCHED_GROUP(0x008, 8, 0);    // MFMA/WMMA group
        }

        if ((s & 7) == 7) {
            // full D sweep done for this N-tile: dist = c2[col] - 2*dot
            const int nt = (s >> 3) * BN;
#pragma unroll
            for (int n = 0; n < 4; ++n) {
                int col = nt + wc * 64 + n * 16 + lo;
                float c2v = c2[col];
#pragma unroll
                for (int m = 0; m < 2; ++m)
#pragma unroll
                    for (int r = 0; r < 8; ++r) {
                        float d = c2v - 2.0f * acc[m][n][r];
                        if (d < minv[m][r]) { minv[m][r] = d; mini[m][r] = col; }
                        acc[m][n][r] = 0.0f;
                    }
            }
        }
    }

    // ---- cross-lane min-reduce within each 16-lane half (row = 8*hi + r) --
#pragma unroll
    for (int m = 0; m < 2; ++m)
#pragma unroll
        for (int r = 0; r < 8; ++r) {
            float v = minv[m][r];
            int   i = mini[m][r];
            for (int mk = 8; mk >= 1; mk >>= 1) {
                float ov = __shfl_xor(v, mk, 32);
                int   oi = __shfl_xor(i, mk, 32);
                if (ov < v || (ov == v && oi < i)) { v = ov; i = oi; }
            }
            minv[m][r] = v;
            mini[m][r] = i;
        }

    // ---- combine the two column-half waves (wc=0 and wc=1) via LDS --------
    float* redv = (float*)(void*)smem;           // 128 floats
    int*   redi = (int*)(void*)(smem + 512);     // 128 ints (bytes 1024..1535)
    __syncthreads();
    if (wc == 1 && lo == 0) {
#pragma unroll
        for (int m = 0; m < 2; ++m)
#pragma unroll
            for (int r = 0; r < 8; ++r) {
                int rl = wr * 32 + m * 16 + hi * 8 + r;
                redv[rl] = minv[m][r];
                redi[rl] = mini[m][r];
            }
    }
    __syncthreads();
    if (wc == 0 && lo == 0) {
#pragma unroll
        for (int m = 0; m < 2; ++m)
#pragma unroll
            for (int r = 0; r < 8; ++r) {
                int rl = wr * 32 + m * 16 + hi * 8 + r;
                float v = minv[m][r];
                int   i = mini[m][r];
                float ov = redv[rl];
                int   oi = redi[rl];
                if (ov < v || (ov == v && oi < i)) { v = ov; i = oi; }
                idx_out[rowBase + rl]  = i;
                idxf_out[rowBase + rl] = (float)i;
            }
    }
}

// ---- gather z_q = codebook[idx], masked squared error, mask count ---------
__global__ __launch_bounds__(256) void vq_gather_loss(
        const float* __restrict__ h,
        const float* __restrict__ cb,
        const int* __restrict__ idx,
        const int* __restrict__ mask,
        float* __restrict__ zq,
        float* accum) {
    __shared__ float red[256];
    int row = blockIdx.x;
    int t = threadIdx.x;
    int ci = idx[row];
    const float* hp = h  + (size_t)row * DDIM + t * 4;
    const float* cp = cb + (size_t)ci  * DDIM + t * 4;
    float4 hv = *(const float4*)hp;
    float4 zv = *(const float4*)cp;
    *(float4*)(zq + (size_t)row * DDIM + t * 4) = zv;   // z_q == z numerically
    float dx = hv.x - zv.x, dy = hv.y - zv.y, dz = hv.z - zv.z, dw = hv.w - zv.w;
    red[t] = dx * dx + dy * dy + dz * dz + dw * dw;
    __syncthreads();
    for (int s = 128; s > 0; s >>= 1) {
        if (t < s) red[t] += red[t + s];
        __syncthreads();
    }
    if (t == 0) {
        float m = (float)mask[row];
        if (m != 0.0f) atomicAdd(&accum[0], red[0] * m);
        atomicAdd(&accum[1], m);
    }
}

// ---- finalize: both losses share the same forward value -------------------
__global__ void vq_finalize(const float* accum, float* losses) {
    float denom = accum[1] * (float)DDIM + 1e-8f;
    float l = accum[0] / denom;
    losses[0] = l;   // commit_loss
    losses[1] = l;   // codebook_loss
}

extern "C" void kernel_launch(void* const* d_in, const int* in_sizes, int n_in,
                              void* d_out, int out_size, void* d_ws, size_t ws_size,
                              hipStream_t stream) {
    (void)in_sizes; (void)n_in; (void)out_size; (void)ws_size;
    const float* h    = (const float*)d_in[0];   // [4,4096,1024]
    const float* cb   = (const float*)d_in[1];   // [8192,1024]
    const int*   mask = (const int*)d_in[2];     // [4,4096]

    // workspace layout
    char* ws = (char*)d_ws;
    unsigned short* hbf  = (unsigned short*)ws;                             // 32 MB
    unsigned short* cbbf = (unsigned short*)(ws + (size_t)BT * DDIM * 2);   // 16 MB
    float* c2     = (float*)(ws + (size_t)BT * DDIM * 2 + (size_t)KCB * DDIM * 2);
    int*   idx_ws = (int*)((char*)c2 + KCB * sizeof(float));
    float* accum  = (float*)((char*)idx_ws + BT * sizeof(int));

    // output layout: z_q | indices(float) | commit | codebook
    float* out    = (float*)d_out;
    float* zq     = out;
    float* idxf   = out + (size_t)BT * DDIM;
    float* losses = idxf + BT;

    vq_init<<<1, 32, 0, stream>>>(accum);
    vq_cvt_h<<<(BT * DDIM) / (256 * 4), 256, 0, stream>>>(h, hbf);
    vq_cvt_cb<<<KCB, 256, 0, stream>>>(cb, cbbf, c2);
    vq_argmin_wmma<<<BT / BM, 256, 2 * (BM + BN) * BK * 2, stream>>>(
        hbf, cbbf, c2, idx_ws, idxf);
    vq_gather_loss<<<BT, 256, 0, stream>>>(h, cb, idx_ws, mask, zq, accum);
    vq_finalize<<<1, 1, 0, stream>>>(accum, losses);
}
